// Seq2Seq_12575664242975
// MI455X (gfx1250) — matrix operand order repaired and checked by reading, not compiled, not run on hardware
//
#include <hip/hip_runtime.h>

typedef __attribute__((ext_vector_type(16))) _Float16 v16h;
typedef __attribute__((ext_vector_type(8)))  _Float16 v8h;
typedef __attribute__((ext_vector_type(8)))  float    v8f;

#define B_TOT 4096
#define T_ENC 168
#define HOR   24
// f16 weight-fragment buffer layout in d_ws (units: _Float16 elements)
#define WENC_OFF 0         // 2 layers * 16 ntiles * 4 kfrags * 512
#define WDEC_OFF 65536
#define WR1_OFF  131072    // 4 ntiles * 2 kfrags * 512
#define WR2_OFF  135168    // 1 ntile  * 2 kfrags * 512
#define WTOTAL   136192    // = 272 KB of d_ws

// ---------------------------------------------------------------------------
// Prep kernel: convert fp32 weights -> f16, pre-swizzled into this file's
// per-lane layout of a WMMA B-fragment (32x16): within a fragment, lane L
// holds 16 contiguous halves; n = nt*16 + (L&15),
// k = kk*32 + (L>=16 ? 16 : 0) + h.  Combined K = [x(0..63) | h(64..127)].
// ---------------------------------------------------------------------------
__global__ void pack_weights(const float* __restrict__ eWih, const float* __restrict__ eWhh,
                             const float* __restrict__ dWih, const float* __restrict__ dWhh,
                             const float* __restrict__ W1,   const float* __restrict__ W2,
                             _Float16* __restrict__ wbuf)
{
  int idx = blockIdx.x * 256 + threadIdx.x;
  if (idx >= WTOTAL) return;
  float val;
  if (idx < WR1_OFF) {
    const float* Wih = (idx < WDEC_OFF) ? eWih : dWih;
    const float* Whh = (idx < WDEC_OFF) ? eWhh : dWhh;
    int e    = idx & 65535;
    int frag = e >> 9, w = e & 511, lane = w >> 4, h = w & 15;
    int l  = frag >> 6, nt = (frag >> 2) & 15, kk = frag & 3;
    int n  = nt * 16 + (lane & 15);
    int k  = kk * 32 + ((lane >> 4) << 4) + h;
    val = (k < 64) ? Wih[(l * 256 + n) * 64 + k]
                   : Whh[(l * 256 + n) * 64 + (k - 64)];
  } else if (idx < WR2_OFF) {
    int e = idx - WR1_OFF;
    int frag = e >> 9, w = e & 511, lane = w >> 4, h = w & 15;
    int nt = frag >> 1, kk = frag & 1;
    int n  = nt * 16 + (lane & 15);
    int k  = kk * 32 + ((lane >> 4) << 4) + h;
    val = W1[n * 64 + k];
  } else {
    int e = idx - WR2_OFF;
    int frag = e >> 9, w = e & 511, lane = w >> 4, h = w & 15;
    int n  = lane & 15;
    int k  = frag * 32 + ((lane >> 4) << 4) + h;
    val = W2[n * 64 + k];
  }
  wbuf[idx] = (_Float16)val;
}

// ---------------------------------------------------------------------------
// Device helpers
// ---------------------------------------------------------------------------
#if __has_builtin(__builtin_amdgcn_tanhf)
__device__ __forceinline__ float ftanh_(float x){ return __builtin_amdgcn_tanhf(x); }
__device__ __forceinline__ float fsig(float x)  { return 0.5f + 0.5f * __builtin_amdgcn_tanhf(0.5f * x); }
#elif __has_builtin(__builtin_amdgcn_tanh_f32)
__device__ __forceinline__ float ftanh_(float x){ return __builtin_amdgcn_tanh_f32(x); }
__device__ __forceinline__ float fsig(float x)  { return 0.5f + 0.5f * __builtin_amdgcn_tanh_f32(0.5f * x); }
#else
__device__ __forceinline__ float fexp2f_(float x){ return __builtin_amdgcn_exp2f(x); }
__device__ __forceinline__ float frcpf_(float x){ return __builtin_amdgcn_rcpf(x); }
__device__ __forceinline__ float fsig(float x)  { return frcpf_(1.0f + fexp2f_(-1.44269504f * x)); }
__device__ __forceinline__ float ftanh_(float x){ return 1.0f - 2.0f * frcpf_(1.0f + fexp2f_(2.88539008f * x)); }
#endif

__device__ __forceinline__ v16h cat16(v8h lo, v8h hi){
  v16h a;
#pragma unroll
  for (int u = 0; u < 8; ++u){ a[u] = lo[u]; a[u + 8] = hi[u]; }
  return a;
}

// A-fragment (16x32 f16) from LDS staging [16][128] f16, row-major.
// Layout: lanes0-15 hold K kk*32+{0..15}; lanes16-31 {16..31} (matches B packing).
__device__ __forceinline__ v16h load_afrag(const _Float16* S, int lane, int kk){
  int base = (lane & 15) * 128 + kk * 32 + (lane >> 4) * 16;
  v8h lo = *(const v8h*)(S + base);
  v8h hi = *(const v8h*)(S + base + 8);
  return cat16(lo, hi);
}

// B-fragment: 32 contiguous bytes per lane from the pre-swizzled buffer.
__device__ __forceinline__ v16h load_bfrag(const _Float16* __restrict__ w, int fragIdx, int lane){
  const v8h* p = (const v8h*)(w + fragIdx * 512 + lane * 16);
  return cat16(p[0], p[1]);
}

__device__ __forceinline__ v8f splat8(float x){
  v8f v;
#pragma unroll
  for (int r = 0; r < 8; ++r) v[r] = x;
  return v;
}

// One LSTM layer step for a 16-row tile. Wave wv owns hidden cols
// wv*16..wv*16+15 of all four gates (n-tiles {g*4+wv}), so the whole
// c/h elementwise update stays in registers.
__device__ __forceinline__ void lstm_layer(const _Float16* Asrc,
                                           const v16h (&W)[4][4],
                                           const float (&bg)[4],
                                           v8f& c, float (&hout)[8], int lane)
{
  v8f a0 = splat8(bg[0]), a1 = splat8(bg[1]), a2 = splat8(bg[2]), a3 = splat8(bg[3]);
#pragma unroll
  for (int kk = 0; kk < 4; ++kk){
    v16h a = load_afrag(Asrc, lane, kk);
    a0 = __builtin_amdgcn_wmma_f32_16x16x32_f16(false, a, false, W[0][kk], (short)0, a0, false, false);
    a1 = __builtin_amdgcn_wmma_f32_16x16x32_f16(false, a, false, W[1][kk], (short)0, a1, false, false);
    a2 = __builtin_amdgcn_wmma_f32_16x16x32_f16(false, a, false, W[2][kk], (short)0, a2, false, false);
    a3 = __builtin_amdgcn_wmma_f32_16x16x32_f16(false, a, false, W[3][kk], (short)0, a3, false, false);
  }
#pragma unroll
  for (int r = 0; r < 8; ++r){
    float iv = fsig(a0[r]);
    float fv = fsig(a1[r]);
    float gv = ftanh_(a2[r]);
    float ov = fsig(a3[r]);
    float cn = fv * c[r] + iv * gv;
    c[r] = cn;
    hout[r] = ov * ftanh_(cn);
  }
}

// Scatter a wave's h tile (C-layout: lane<16 row=r, lane>=16 row=8+r,
// col = wv*16 + (lane&15)) into an LDS staging buffer as f16.
__device__ __forceinline__ void store_h(_Float16* S, const float (&h)[8], int lane, int wv, int off){
  int col = off + wv * 16 + (lane & 15);
  int rb  = (lane >> 4) * 8;
#pragma unroll
  for (int r = 0; r < 8; ++r) S[(rb + r) * 128 + col] = (_Float16)h[r];
}

// ---------------------------------------------------------------------------
// Persistent seq2seq kernel: one 16-row batch tile per workgroup,
// 4 waves, whole encoder+decoder+readout in one launch.
// ---------------------------------------------------------------------------
__global__ __launch_bounds__(128) void seq2seq_main(
    const float* __restrict__ X,   const float* __restrict__ h0,
    const float* __restrict__ c0e, const float* __restrict__ c0d,
    const float* __restrict__ encB, const float* __restrict__ decB,
    const float* __restrict__ rb1,  const float* __restrict__ rb2,
    const _Float16* __restrict__ wbuf, float* __restrict__ out)
{
  __shared__ _Float16 A0[16 * 128];   // [:, :64]=layer0 input, [:,64:]=h0 state
  __shared__ _Float16 A1[16 * 128];   // [:, :64]=layer1 input, [:,64:]=h1 state

  const int tid   = threadIdx.x;
  const int lane  = tid & 31;
  const int wv    = tid >> 5;
  const int bbase = blockIdx.x * 16;
  const int xrow  = tid >> 3;         // staging role: row 0..15
  const int xcol  = (tid & 7) * 8;    // staging role: 8-wide column chunk

  // ---- prologue: stage x_0 and h0 (both layers) as f16 ----
  {
    const float* xp = X + ((bbase + xrow) * T_ENC + 0) * 64 + xcol;
    v8h hx;
#pragma unroll
    for (int u = 0; u < 8; ++u) hx[u] = (_Float16)xp[u];
    *(v8h*)&A0[xrow * 128 + xcol] = hx;

    const float* hp0 = h0 + (0 * B_TOT + bbase + xrow) * 64 + xcol;
    const float* hp1 = h0 + (1 * B_TOT + bbase + xrow) * 64 + xcol;
    v8h h0x, h1x;
#pragma unroll
    for (int u = 0; u < 8; ++u){ h0x[u] = (_Float16)hp0[u]; h1x[u] = (_Float16)hp1[u]; }
    *(v8h*)&A0[xrow * 128 + 64 + xcol] = h0x;
    *(v8h*)&A1[xrow * 128 + 64 + xcol] = h1x;
  }

  // ---- encoder weights (register resident), biases, c state ----
  v16h W[2][4][4];
#pragma unroll
  for (int l = 0; l < 2; ++l)
#pragma unroll
    for (int g = 0; g < 4; ++g)
#pragma unroll
      for (int kk = 0; kk < 4; ++kk)
        W[l][g][kk] = load_bfrag(wbuf + WENC_OFF, (l * 16 + (g * 4 + wv)) * 4 + kk, lane);

  float bg[2][4];
#pragma unroll
  for (int l = 0; l < 2; ++l)
#pragma unroll
    for (int g = 0; g < 4; ++g)
      bg[l][g] = encB[l * 256 + g * 64 + wv * 16 + (lane & 15)];

  v8f c0r, c1r;
  {
    int col = wv * 16 + (lane & 15);
    int rb  = (lane >> 4) * 8;
#pragma unroll
    for (int r = 0; r < 8; ++r){
      c0r[r] = c0e[(0 * B_TOT + bbase + rb + r) * 64 + col];
      c1r[r] = c0e[(1 * B_TOT + bbase + rb + r) * 64 + col];
    }
  }
  __syncthreads();

  float h0o[8], h1o[8];

  // ---- encoder: 168 sequential steps, x(t+1) prefetched during compute ----
  // 3 barriers/step: B1 (A0 reads done -> A0/A1 stores), B2 (stores visible
  // -> layer1 reads), B3 (A1 reads done -> h1 state store). The h1 store is
  // consumed only after the *next* step's B2, so no trailing barrier needed.
  for (int t = 0; t < T_ENC; ++t){
    float xr[8];
    const bool havex = (t + 1 < T_ENC);
    if (havex){
      const float* xp = X + ((bbase + xrow) * T_ENC + (t + 1)) * 64 + xcol;
#pragma unroll
      for (int u = 0; u < 8; ++u) xr[u] = xp[u];
    }

    lstm_layer(A0, W[0], bg[0], c0r, h0o, lane);
    __syncthreads();                       // B1: all waves done reading A0
    store_h(A0, h0o, lane, wv, 64);        // persist h0 state
    store_h(A1, h0o, lane, wv, 0);         // feed layer 1
    if (havex){
      v8h hx;
#pragma unroll
      for (int u = 0; u < 8; ++u) hx[u] = (_Float16)xr[u];
      *(v8h*)&A0[xrow * 128 + xcol] = hx;  // next step's x
    }
    __syncthreads();                       // B2
    lstm_layer(A1, W[1], bg[1], c1r, h1o, lane);
    __syncthreads();                       // B3: all waves done reading A1
    store_h(A1, h1o, lane, wv, 64);        // persist h1 state (no barrier)
  }
  __syncthreads();                         // one-time: h1 stores visible below

  // ---- swap in decoder weights / biases / c, readout weights ----
#pragma unroll
  for (int l = 0; l < 2; ++l)
#pragma unroll
    for (int g = 0; g < 4; ++g)
#pragma unroll
      for (int kk = 0; kk < 4; ++kk)
        W[l][g][kk] = load_bfrag(wbuf + WDEC_OFF, (l * 16 + (g * 4 + wv)) * 4 + kk, lane);
#pragma unroll
  for (int l = 0; l < 2; ++l)
#pragma unroll
    for (int g = 0; g < 4; ++g)
      bg[l][g] = decB[l * 256 + g * 64 + wv * 16 + (lane & 15)];
  {
    int col = wv * 16 + (lane & 15);
    int rb  = (lane >> 4) * 8;
#pragma unroll
    for (int r = 0; r < 8; ++r){
      c0r[r] = c0d[(0 * B_TOT + bbase + rb + r) * 64 + col];
      c1r[r] = c0d[(1 * B_TOT + bbase + rb + r) * 64 + col];
    }
  }
  v16h Wr1[2], Wr2[2];
#pragma unroll
  for (int kk = 0; kk < 2; ++kk){
    Wr1[kk] = load_bfrag(wbuf + WR1_OFF, wv * 2 + kk, lane);
    Wr2[kk] = load_bfrag(wbuf + WR2_OFF, kk, lane);
  }
  const float b1v = rb1[wv * 16 + (lane & 15)];
  const float b2v = rb2[lane & 15];

  // decoder first input = encoder top-layer h (in A1[:,64:128])
  {
    v8h tp = *(v8h*)&A1[xrow * 128 + 64 + xcol];
    *(v8h*)&A0[xrow * 128 + xcol] = tp;
  }
  __syncthreads();

  // ---- decoder: 24 steps + fused readout MLP (5 barriers/step) ----
  for (int s = 0; s < HOR; ++s){
    lstm_layer(A0, W[0], bg[0], c0r, h0o, lane);
    __syncthreads();                       // B1
    store_h(A0, h0o, lane, wv, 64);
    store_h(A1, h0o, lane, wv, 0);
    __syncthreads();                       // B2
    lstm_layer(A1, W[1], bg[1], c1r, h1o, lane);
    __syncthreads();                       // B3
    store_h(A1, h1o, lane, wv, 64);
    store_h(A0, h1o, lane, wv, 0);         // next input AND readout feats
    __syncthreads();                       // B4: feats visible

    // hidden = relu(feats @ W1^T + b1): wave wv computes hidden cols wv*16..+15
    v8f ha = splat8(b1v);
#pragma unroll
    for (int kk = 0; kk < 2; ++kk){
      v16h a = load_afrag(A0, lane, kk);
      ha = __builtin_amdgcn_wmma_f32_16x16x32_f16(false, a, false, Wr1[kk], (short)0, ha, false, false);
    }
    float hr[8];
#pragma unroll
    for (int r = 0; r < 8; ++r) hr[r] = ha[r] > 0.0f ? ha[r] : 0.0f;
    store_h(A1, hr, lane, wv, 0);          // A1[:, :64] free since B3
    __syncthreads();                       // B5: hidden visible

    if (wv == 0){                          // out tile is a single 16x16 WMMA
      v8f oa = splat8(b2v);
#pragma unroll
      for (int kk = 0; kk < 2; ++kk){
        v16h a = load_afrag(A1, lane, kk);
        oa = __builtin_amdgcn_wmma_f32_16x16x32_f16(false, a, false, Wr2[kk], (short)0, oa, false, false);
      }
      int col = lane & 15;
      int rb  = (lane >> 4) * 8;
#pragma unroll
      for (int r = 0; r < 8; ++r)
        out[((bbase + rb + r) * HOR + s) * 16 + col] = oa[r];
    }
    // no tail barrier: waves 1-3 start next step's layer-0 WMMAs (A0 is
    // stable); wave 0's A1 reads complete before it joins next B1.
  }
}

// ---------------------------------------------------------------------------
extern "C" void kernel_launch(void* const* d_in, const int* in_sizes, int n_in,
                              void* d_out, int out_size, void* d_ws, size_t ws_size,
                              hipStream_t stream)
{
  const float* X    = (const float*)d_in[0];
  // d_in[1] = X_mask: unused by the reference as well
  const float* h0   = (const float*)d_in[2];
  const float* c0e  = (const float*)d_in[3];
  const float* c0d  = (const float*)d_in[4];
  const float* eWih = (const float*)d_in[5];
  const float* eWhh = (const float*)d_in[6];
  const float* eB   = (const float*)d_in[7];
  const float* dWih = (const float*)d_in[8];
  const float* dWhh = (const float*)d_in[9];
  const float* dB   = (const float*)d_in[10];
  const float* W1   = (const float*)d_in[11];
  const float* b1   = (const float*)d_in[12];
  const float* W2   = (const float*)d_in[13];
  const float* b2   = (const float*)d_in[14];

  _Float16* wbuf = (_Float16*)d_ws;        // needs 272 KB of scratch

  pack_weights<<<(WTOTAL + 255) / 256, 256, 0, stream>>>(eWih, eWhh, dWih, dWhh, W1, W2, wbuf);
  seq2seq_main<<<B_TOT / 16, 128, 0, stream>>>(X, h0, c0e, c0d, eB, dB, b1, b2, wbuf, (float*)d_out);
}
